// CEVAE_43130061587068
// MI455X (gfx1250) — compile-verified, matrix operand order checked by reading, not run on hardware
//
#include <hip/hip_runtime.h>
#include <stdint.h>

// ---------------------------------------------------------------------------
// CDNA5 / gfx1250 CEVAE forward:
//   pass 1: convert x + all weights to bf16 (weights transposed to [N][K])
//   pass 2: bf16 WMMA GEMMs with async global->LDS staging (ASYNCcnt pipeline)
//   heads : wave-per-row VALU kernels
// ---------------------------------------------------------------------------

typedef __attribute__((ext_vector_type(8)))  float  v8f;
typedef __attribute__((ext_vector_type(16))) __bf16 v16bf;

#define BROWS 16384
#define FDIM  1024
#define HDIM  1024
#define LDIM  256
#define KTRT  7
#define OUTW  523   // 7 + 4 + 512

#define BM 128
#define BN 128
#define BK 32
#define LDAS 40     // padded LDS row stride (elements): 80B, 16B-aligned rows

__device__ __forceinline__ uint16_t f2bf(float f) {
    uint32_t u = __float_as_uint(f);
    u = (u + 0x7FFFu + ((u >> 16) & 1u)) >> 16;   // round-to-nearest-even
    return (uint16_t)u;
}
__device__ __forceinline__ float bf2f(uint16_t h) {
    return __uint_as_float(((uint32_t)h) << 16);
}
__device__ __forceinline__ float eluf(float x)      { return x > 0.f ? x : (expf(x) - 1.f); }
__device__ __forceinline__ float softplusf(float x) { return x > 20.f ? x : log1pf(expf(x)); }
__device__ __forceinline__ uint32_t pk2(uint16_t a, uint16_t b) {
    return (uint32_t)a | ((uint32_t)b << 16);
}

union FragBF { uint4 u[2]; v16bf v; };

// ---------------------------------------------------------------------------
// elementwise f32 -> bf16 (4 elements / thread)
// ---------------------------------------------------------------------------
__global__ __launch_bounds__(256)
void cvt_bf16_k(const float* __restrict__ src, uint16_t* __restrict__ dst)
{
    size_t i = ((size_t)blockIdx.x * 256 + threadIdx.x) * 4;
    float4 f = *(const float4*)(src + i);
    uint2 p;
    p.x = pk2(f2bf(f.x), f2bf(f.y));
    p.y = pk2(f2bf(f.z), f2bf(f.w));
    *(uint2*)(dst + i) = p;
}

// ---------------------------------------------------------------------------
// W [Kd][N] f32  ->  Wt [N][Kd] bf16   (32x32 LDS tiles; blockIdx.z batches)
// ---------------------------------------------------------------------------
__global__ __launch_bounds__(256)
void transp_cvt_k(const float* __restrict__ src, uint16_t* __restrict__ dst,
                  int Kd, int N)
{
    __shared__ uint16_t tile[32][33];
    const int tid = threadIdx.x;
    src += (size_t)blockIdx.z * Kd * N;
    dst += (size_t)blockIdx.z * N * Kd;
    const int k0 = blockIdx.y * 32, n0 = blockIdx.x * 32;
#pragma unroll
    for (int i = 0; i < 4; ++i) {
        int idx = tid + i * 256;          // 0..1023
        int kl = idx >> 5, nl = idx & 31;
        tile[kl][nl] = f2bf(src[(size_t)(k0 + kl) * N + n0 + nl]);
    }
    __syncthreads();
#pragma unroll
    for (int i = 0; i < 4; ++i) {
        int idx = tid + i * 256;
        int nl = idx >> 5, kl = idx & 31;
        dst[(size_t)(n0 + nl) * Kd + k0 + kl] = tile[kl][nl];
    }
}

// ---------------------------------------------------------------------------
// Tiled bf16 WMMA GEMM:  C[b,n] = sum_k A[b,k] * Wt[n,k]   (f32 accumulate)
// A: bf16 [M][Kdim] row-major. Wt: bf16 [N][Kdim] row-major (pre-transposed).
// Staging: global_load_async_to_lds_b128, double-buffered, ASYNCcnt pipelined.
//   EPI 0: out_bf16 = elu(acc + bias[n])
//   EPI 1: out_bf16 = elu(acc + bias[n] + y[b]*wy[n] + d[b]*wd[n])   (z layer 1)
//   EPI 2: z-head, gridDim.z = K: Wt -> Wt + z*N*Kdim, bias -> bias + z*N,
//          store f32 to out[b, 11+n] only where tsel[b]==blockIdx.z
// ---------------------------------------------------------------------------
template<int EPI>
__global__ __launch_bounds__(256)
void gemm_wmma_k(const uint16_t* __restrict__ A,
                 const uint16_t* __restrict__ Wt,
                 const float*    __restrict__ BiasB,
                 uint16_t*       __restrict__ OutBf,
                 float*          __restrict__ OutF,
                 const float*    __restrict__ yvec,
                 const float*    __restrict__ dvec,
                 const float*    __restrict__ wy,
                 const float*    __restrict__ wd,
                 const int*      __restrict__ tsel,
                 int N, int Kdim, int lda)
{
    __shared__ alignas(16) uint16_t lsA[2][BM * LDAS];
    __shared__ alignas(16) uint16_t lsB[2][BN * LDAS];

    const int tid = threadIdx.x;
    const int bm  = blockIdx.y * BM;
    const int bn  = blockIdx.x * BN;
    const int kz  = blockIdx.z;
    const uint16_t* __restrict__ Wk   = Wt    + (size_t)kz * (size_t)Kdim * (size_t)N;
    const float*    __restrict__ Bias = BiasB + (size_t)kz * (size_t)N;

    const int wid  = tid >> 5;
    const int lane = tid & 31;
    const int half = lane >> 4;       // element->K mapping half
    const int lrow = lane & 15;
    const int Moff = (wid >> 1) * 32; // 4 wave-rows x 2 wave-cols
    const int Noff = (wid & 1) * 64;

    // staging geometry: 128x32 bf16 tile = 512 x b128; 256 thr -> 2 each
    const int sm0 = tid >> 2;          // 0..63  (i=0) ; +64 for i=1
    const int skq = (tid & 3) * 8;     // element offset 0,8,16,24

    v8f acc[2][4];
#pragma unroll
    for (int i = 0; i < 2; ++i)
#pragma unroll
        for (int j = 0; j < 4; ++j) acc[i][j] = (v8f)(0.0f);

    // issue one pipeline stage (4 async b128 loads per thread, 4 per wave)
    auto issue = [&](int kt, int s) {
#pragma unroll
        for (int i = 0; i < 2; ++i) {
            const int m = sm0 + i * 64;                 // 0..127
            uint32_t la = (uint32_t)(uintptr_t)&lsA[s][m * LDAS + skq];
            uint64_t ga = (uint64_t)(uintptr_t)(A + (size_t)(bm + m) * lda + kt + skq);
            asm volatile("global_load_async_to_lds_b128 %0, %1, off"
                         :: "v"(la), "v"(ga) : "memory");
            uint32_t lb = (uint32_t)(uintptr_t)&lsB[s][m * LDAS + skq];
            uint64_t gb = (uint64_t)(uintptr_t)(Wk + (size_t)(bn + m) * Kdim + kt + skq);
            asm volatile("global_load_async_to_lds_b128 %0, %1, off"
                         :: "v"(lb), "v"(gb) : "memory");
        }
    };

    issue(0, 0);
    int s = 0;
    for (int kt = 0; kt < Kdim; kt += BK) {
        if (kt + BK < Kdim) {
            issue(kt + BK, s ^ 1);
            asm volatile("s_wait_asynccnt 0x4" ::: "memory"); // retire stage kt
        } else {
            asm volatile("s_wait_asynccnt 0x0" ::: "memory");
        }
        __syncthreads();

        FragBF afr[2], bfr[4];
#pragma unroll
        for (int ma = 0; ma < 2; ++ma) {
            const uint16_t* p = &lsA[s][(Moff + ma * 16 + lrow) * LDAS];
            afr[ma].u[0] = *(const uint4*)(p + 8 * half);
            afr[ma].u[1] = *(const uint4*)(p + 16 + 8 * half);
        }
#pragma unroll
        for (int nb = 0; nb < 4; ++nb) {
            const uint16_t* p = &lsB[s][(Noff + nb * 16 + lrow) * LDAS];
            bfr[nb].u[0] = *(const uint4*)(p + 8 * half);
            bfr[nb].u[1] = *(const uint4*)(p + 16 + 8 * half);
        }
#pragma unroll
        for (int ma = 0; ma < 2; ++ma)
#pragma unroll
            for (int nb = 0; nb < 4; ++nb)
                acc[ma][nb] = __builtin_amdgcn_wmma_f32_16x16x32_bf16(
                    false, afr[ma].v, false, bfr[nb].v,
                    (short)0, acc[ma][nb], false, false);
        __syncthreads();   // protect buffer s from the stage kt+2 issue
        s ^= 1;
    }

    // ---- epilogue ----
#pragma unroll
    for (int ma = 0; ma < 2; ++ma) {
#pragma unroll
        for (int nb = 0; nb < 4; ++nb) {
            const int gn = bn + Noff + nb * 16 + lrow;
#pragma unroll
            for (int r = 0; r < 8; ++r) {
                const int gm = bm + Moff + ma * 16 + half * 8 + r;
                float v = acc[ma][nb][r];
                if constexpr (EPI == 0) {
                    v = eluf(v + Bias[gn]);
                    OutBf[(size_t)gm * N + gn] = f2bf(v);
                } else if constexpr (EPI == 1) {
                    v += Bias[gn] + yvec[gm] * wy[gn] + dvec[gm] * wd[gn];
                    v = eluf(v);
                    OutBf[(size_t)gm * N + gn] = f2bf(v);
                } else {
                    if (tsel[gm] == kz) {
                        v += Bias[gn];
                        if (gn < LDIM) v = fminf(fmaxf(v, -100.f), 100.f);      // z_loc
                        else           v = fminf(softplusf(v) + 0.001f, 100.f); // z_scale
                        OutF[(size_t)gm * OUTW + 11 + gn] = v;
                    }
                }
            }
        }
    }
}

// ---------------------------------------------------------------------------
// t-final: logits = clip(elu(h2t[B,256] @ tw2[256,7] + tb2), -10, 10)
// ---------------------------------------------------------------------------
__global__ __launch_bounds__(256)
void tfinal_k(const uint16_t* __restrict__ h2, const float* __restrict__ tw2,
              const float* __restrict__ tb2, float* __restrict__ out)
{
    __shared__ float lw[LDIM * KTRT];
    const int tid = threadIdx.x;
    for (int i = tid; i < LDIM * KTRT; i += 256) lw[i] = tw2[i];
    __syncthreads();
    const int wid = tid >> 5, lane = tid & 31;
    const int row = blockIdx.x * 8 + wid;
    float acc[KTRT] = {0.f, 0.f, 0.f, 0.f, 0.f, 0.f, 0.f};
    for (int j = lane; j < LDIM; j += 32) {
        float a = bf2f(h2[(size_t)row * LDIM + j]);
        const float* wr = &lw[j * KTRT];
#pragma unroll
        for (int c = 0; c < KTRT; ++c) acc[c] += a * wr[c];
    }
#pragma unroll
    for (int c = 0; c < KTRT; ++c) {
        float sv = acc[c];
        for (int o = 16; o; o >>= 1) sv += __shfl_xor(sv, o, 32);
        acc[c] = sv;
    }
    if (lane == 0) {
#pragma unroll
        for (int c = 0; c < KTRT; ++c) {
            float v = eluf(acc[c] + tb2[c]);
            out[(size_t)row * OUTW + c] = fminf(fmaxf(v, -10.f), 10.f);
        }
    }
}

// ---------------------------------------------------------------------------
// y/d head: sel = h2[row] @ hW[t[row]] + hb[t[row]]  -> (loc, scale)
// ---------------------------------------------------------------------------
__global__ __launch_bounds__(256)
void head_k(const uint16_t* __restrict__ h2, const float* __restrict__ hW,
            const float* __restrict__ hb, const int* __restrict__ tsel,
            float* __restrict__ out, int col0)
{
    const int tid = threadIdx.x, wid = tid >> 5, lane = tid & 31;
    const int row = blockIdx.x * 8 + wid;
    const int tk  = tsel[row];
    const float* Wp = hW + (size_t)tk * HDIM * 2;
    float a0 = 0.f, a1 = 0.f;
    for (int j = lane; j < HDIM; j += 32) {
        float a = bf2f(h2[(size_t)row * HDIM + j]);
        a0 += a * Wp[2 * j];
        a1 += a * Wp[2 * j + 1];
    }
    for (int o = 16; o; o >>= 1) { a0 += __shfl_xor(a0, o, 32); a1 += __shfl_xor(a1, o, 32); }
    if (lane == 0) {
        float loc = fminf(fmaxf(a0 + hb[2 * tk], -1e6f), 1e6f);
        float sc  = fminf(softplusf(a1 + hb[2 * tk + 1]) + 1e-3f, 1e6f);
        out[(size_t)row * OUTW + col0]     = loc;
        out[(size_t)row * OUTW + col0 + 1] = sc;
    }
}

// ---------------------------------------------------------------------------
extern "C" void kernel_launch(void* const* d_in, const int* in_sizes, int n_in,
                              void* d_out, int out_size, void* d_ws, size_t ws_size,
                              hipStream_t stream) {
    (void)in_sizes; (void)n_in; (void)out_size; (void)ws_size;
    const float* x   = (const float*)d_in[0];
    const int*   t   = (const int*)  d_in[1];
    const float* y   = (const float*)d_in[2];
    const float* dd  = (const float*)d_in[3];
    const float* tw0 = (const float*)d_in[4];  const float* tb0 = (const float*)d_in[5];
    const float* tw1 = (const float*)d_in[6];  const float* tb1 = (const float*)d_in[7];
    const float* tw2 = (const float*)d_in[8];  const float* tb2 = (const float*)d_in[9];
    const float* yw0 = (const float*)d_in[10]; const float* yb0 = (const float*)d_in[11];
    const float* yw1 = (const float*)d_in[12]; const float* yb1 = (const float*)d_in[13];
    const float* yhW = (const float*)d_in[14]; const float* yhb = (const float*)d_in[15];
    const float* dw0 = (const float*)d_in[16]; const float* db0 = (const float*)d_in[17];
    const float* dw1 = (const float*)d_in[18]; const float* db1 = (const float*)d_in[19];
    const float* dhW = (const float*)d_in[20]; const float* dhb = (const float*)d_in[21];
    const float* zw0 = (const float*)d_in[22]; const float* zb0 = (const float*)d_in[23];
    const float* zw1 = (const float*)d_in[24]; const float* zb1 = (const float*)d_in[25];
    const float* zhW = (const float*)d_in[26]; const float* zhb = (const float*)d_in[27];
    float* out = (float*)d_out;

    // ---- workspace layout (bf16 elements) ----
    uint16_t* p    = (uint16_t*)d_ws;
    uint16_t* hA   = p; p += (size_t)BROWS * HDIM;      // 32 MB
    uint16_t* hB   = p; p += (size_t)BROWS * HDIM;      // 32 MB
    uint16_t* xb   = p; p += (size_t)BROWS * FDIM;      // 32 MB
    uint16_t* tw0t = p; p += (size_t)LDIM * FDIM;
    uint16_t* tw1t = p; p += (size_t)LDIM * LDIM;
    uint16_t* yw0t = p; p += (size_t)HDIM * FDIM;
    uint16_t* yw1t = p; p += (size_t)HDIM * HDIM;
    uint16_t* dw0t = p; p += (size_t)HDIM * FDIM;
    uint16_t* dw1t = p; p += (size_t)HDIM * HDIM;
    uint16_t* zw0t = p; p += (size_t)HDIM * FDIM;
    uint16_t* zw1t = p; p += (size_t)HDIM * HDIM;
    uint16_t* zhWt = p; p += (size_t)KTRT * (2 * LDIM) * HDIM;

    const dim3 blk(256, 1, 1);

    // ---- pass 1: conversions (bandwidth-trivial, ~50 MB total) ----
    cvt_bf16_k<<<(BROWS * FDIM) / 1024, blk, 0, stream>>>(x, xb);
    transp_cvt_k<<<dim3(LDIM / 32, FDIM / 32, 1), blk, 0, stream>>>(tw0, tw0t, FDIM, LDIM);
    transp_cvt_k<<<dim3(LDIM / 32, LDIM / 32, 1), blk, 0, stream>>>(tw1, tw1t, LDIM, LDIM);
    transp_cvt_k<<<dim3(HDIM / 32, FDIM / 32, 1), blk, 0, stream>>>(yw0, yw0t, FDIM, HDIM);
    transp_cvt_k<<<dim3(HDIM / 32, HDIM / 32, 1), blk, 0, stream>>>(yw1, yw1t, HDIM, HDIM);
    transp_cvt_k<<<dim3(HDIM / 32, FDIM / 32, 1), blk, 0, stream>>>(dw0, dw0t, FDIM, HDIM);
    transp_cvt_k<<<dim3(HDIM / 32, HDIM / 32, 1), blk, 0, stream>>>(dw1, dw1t, HDIM, HDIM);
    transp_cvt_k<<<dim3(HDIM / 32, FDIM / 32, 1), blk, 0, stream>>>(zw0 + 2 * HDIM, zw0t, FDIM, HDIM);
    transp_cvt_k<<<dim3(HDIM / 32, HDIM / 32, 1), blk, 0, stream>>>(zw1, zw1t, HDIM, HDIM);
    transp_cvt_k<<<dim3((2 * LDIM) / 32, HDIM / 32, KTRT), blk, 0, stream>>>(zhW, zhWt, HDIM, 2 * LDIM);

    const dim3 gH(HDIM / BN, BROWS / BM, 1);          // N=1024
    const dim3 gL(LDIM / BN, BROWS / BM, 1);          // N=256
    const dim3 gZ((2 * LDIM) / BN, BROWS / BM, KTRT); // N=512, z = treatment

    // ---- t chain ----
    gemm_wmma_k<0><<<gL, blk, 0, stream>>>(xb, tw0t, tb0, hA, nullptr,
        nullptr, nullptr, nullptr, nullptr, nullptr, LDIM, FDIM, FDIM);
    gemm_wmma_k<0><<<gL, blk, 0, stream>>>(hA, tw1t, tb1, hB, nullptr,
        nullptr, nullptr, nullptr, nullptr, nullptr, LDIM, LDIM, LDIM);
    tfinal_k<<<BROWS / 8, blk, 0, stream>>>(hB, tw2, tb2, out);

    // ---- y chain ----
    gemm_wmma_k<0><<<gH, blk, 0, stream>>>(xb, yw0t, yb0, hA, nullptr,
        nullptr, nullptr, nullptr, nullptr, nullptr, HDIM, FDIM, FDIM);
    gemm_wmma_k<0><<<gH, blk, 0, stream>>>(hA, yw1t, yb1, hB, nullptr,
        nullptr, nullptr, nullptr, nullptr, nullptr, HDIM, HDIM, HDIM);
    head_k<<<BROWS / 8, blk, 0, stream>>>(hB, yhW, yhb, t, out, 7);

    // ---- d chain ----
    gemm_wmma_k<0><<<gH, blk, 0, stream>>>(xb, dw0t, db0, hA, nullptr,
        nullptr, nullptr, nullptr, nullptr, nullptr, HDIM, FDIM, FDIM);
    gemm_wmma_k<0><<<gH, blk, 0, stream>>>(hA, dw1t, db1, hB, nullptr,
        nullptr, nullptr, nullptr, nullptr, nullptr, HDIM, HDIM, HDIM);
    head_k<<<BROWS / 8, blk, 0, stream>>>(hB, dhW, dhb, t, out, 9);

    // ---- z chain: layer1 folds y/d rank-1 terms (zw0 rows 0,1) into epilogue ----
    gemm_wmma_k<1><<<gH, blk, 0, stream>>>(xb, zw0t, zb0, hA, nullptr,
        y, dd, zw0, zw0 + HDIM, nullptr, HDIM, FDIM, FDIM);
    gemm_wmma_k<0><<<gH, blk, 0, stream>>>(hA, zw1t, zb1, hB, nullptr,
        nullptr, nullptr, nullptr, nullptr, nullptr, HDIM, HDIM, HDIM);
    // z head: 7 dense GEMMs via gridDim.z, predicated store where t[row]==k
    gemm_wmma_k<2><<<gZ, blk, 0, stream>>>(hB, zhWt, zhb, nullptr, out,
        nullptr, nullptr, nullptr, nullptr, t, 2 * LDIM, HDIM, HDIM);
}
